// SecureFeedForward_70214125355099
// MI455X (gfx1250) — compile-verified, matrix operand order checked by reading, not run on hardware
//
#include <hip/hip_runtime.h>
#include <hip/hip_bf16.h>
#include <cstddef>

// ---------------------------------------------------------------------------
// SecureFeedForward on MI455X (gfx1250, wave32, WMMA).
//   Y = relu(X @ W1^T + b1) @ W2^T + b2
//   X:[M,1024] f32, W1:[4096,1024] f32, W2:[1024,4096] f32, Y:[M,1024] f32
// Compute-bound (2.75e11 FLOP vs ~20us of HBM traffic at 23.3 TB/s) ->
// bf16 WMMA path with fp32 accumulate. Intermediate activations bf16 in d_ws
// (M*4096*2 = 128 MB at M=16384; X/W1/act working sets fit the 192 MB L2).
// Two-stage software pipeline: global->VGPR loads for chunk k+1 are issued
// between the barrier and the WMMA block of chunk k; LDS tiles are
// double-buffered so staging never overlaps readers.
// ---------------------------------------------------------------------------

#define HIDDEN 1024
#define INTER  4096

typedef __attribute__((ext_vector_type(16))) __bf16 v16bf;
typedef __attribute__((ext_vector_type(8)))  float  v8f;

// 256 threads = 8 waves; block tile 128x128; wave tile 64x32 (8 v8f accums).
#define BM 128
#define BN 128
#define BK 64
// LDS row pitch: 72 bf16 = 144 B = 36 banks -> 16 consecutive rows map to 16
// distinct bank groups for the b128 fragment reads; 144 = 9*16 keeps every
// fragment chunk 16-byte aligned.
#define LDS_PITCH_B 144
#define TILE_BYTES (BM * LDS_PITCH_B)

__device__ __forceinline__ unsigned short f32_to_bf16_rne(float f) {
    unsigned int u = __float_as_uint(f);
    u = u + 0x7FFFu + ((u >> 16) & 1u);
    return (unsigned short)(u >> 16);
}

__device__ __forceinline__ unsigned int pack_bf16x2(float lo, float hi) {
    return ((unsigned int)f32_to_bf16_rne(hi) << 16) | f32_to_bf16_rne(lo);
}

// 16x32 bf16 fragment from a row-major [rows][BK] bf16 LDS tile.
// ISA layout (CDNA5 ISA): lanes 0-15 hold K 0..7 then 16..23,
// lanes 16-31 hold K 8..15 then 24..31 -> two 16-B ds_load_b128 per lane.
__device__ __forceinline__ v16bf frag_ld(const char* tile, int row, int kk, int lane) {
    const int kbyte = (kk + (((lane >> 4) & 1) << 3)) * 2;
    const char* p = tile + row * LDS_PITCH_B + kbyte;
    union { v16bf v; uint4 q[2]; } f;
    f.q[0] = *(const uint4*)(p);
    f.q[1] = *(const uint4*)(p + 32);
    return f.v;
}

// ---- pipeline stage 1: global -> VGPR ----
__device__ __forceinline__ void gload_f32(const float* __restrict__ g, int rowBase,
                                          int lda, int k0, float4* r, int tid, bool pf) {
#pragma unroll
    for (int i = 0; i < 8; ++i) {            // 2048 float4 / 256 threads
        int slot = tid + 256 * i;
        int row  = slot >> 4;                // 16 float4 per 64-wide row
        int cq   = slot & 15;
        const float* src = g + (size_t)(rowBase + row) * lda + k0 + cq * 4;
        r[i] = *(const float4*)src;
        if (pf) __builtin_prefetch(src + BK, 0, 1);   // chunk after next
    }
}

__device__ __forceinline__ void gload_bf16(const unsigned short* __restrict__ g,
                                           int rowBase, int lda, int k0, uint4* r,
                                           int tid, bool pf) {
#pragma unroll
    for (int i = 0; i < 4; ++i) {            // 1024 uint4 / 256 threads
        int slot = tid + 256 * i;
        int row  = slot >> 3;                // 8 uint4 per 64-wide row
        int co   = slot & 7;
        const unsigned short* src = g + (size_t)(rowBase + row) * lda + k0 + co * 8;
        r[i] = *(const uint4*)src;
        if (pf) __builtin_prefetch(src + BK, 0, 1);
    }
}

// ---- pipeline stage 2: VGPR -> LDS (with fp32->bf16 conversion) ----
__device__ __forceinline__ void lstore_f32(const float4* r, char* tile, int tid) {
#pragma unroll
    for (int i = 0; i < 8; ++i) {
        int slot = tid + 256 * i;
        int row  = slot >> 4;
        int cq   = slot & 15;
        uint2 pk;
        pk.x = pack_bf16x2(r[i].x, r[i].y);
        pk.y = pack_bf16x2(r[i].z, r[i].w);
        *(uint2*)(tile + row * LDS_PITCH_B + cq * 8) = pk;
    }
}

__device__ __forceinline__ void lstore_bf16(const uint4* r, char* tile, int tid) {
#pragma unroll
    for (int i = 0; i < 4; ++i) {
        int slot = tid + 256 * i;
        int row  = slot >> 3;
        int co   = slot & 7;
        *(uint4*)(tile + row * LDS_PITCH_B + co * 16) = r[i];
    }
}

// ---- shared WMMA compute over one BKx(BM,BN) LDS chunk ----
__device__ __forceinline__ void mma_chunk(const char* tA, const char* tB,
                                          int wm, int wn, int lane, v8f acc[4][2]) {
#pragma unroll
    for (int kk = 0; kk < BK; kk += 32) {
        v16bf a[4], b[2];
#pragma unroll
        for (int i = 0; i < 4; ++i)
            a[i] = frag_ld(tA, wm + i * 16 + (lane & 15), kk, lane);
#pragma unroll
        for (int j = 0; j < 2; ++j)
            b[j] = frag_ld(tB, wn + j * 16 + (lane & 15), kk, lane);
#pragma unroll
        for (int i = 0; i < 4; ++i)
#pragma unroll
            for (int j = 0; j < 2; ++j)
                acc[i][j] = __builtin_amdgcn_wmma_f32_16x16x32_bf16(
                    false, a[i], false, b[j], (short)0, acc[i][j], false, false);
    }
}

// ---------------- GEMM1: act = relu(X @ W1^T + b1), bf16 out ----------------
__global__ __launch_bounds__(256) void ffn_gemm1(const float* __restrict__ X,
                                                 const float* __restrict__ W1,
                                                 const float* __restrict__ b1,
                                                 unsigned short* __restrict__ act) {
    __shared__ __align__(16) char sA[2][TILE_BYTES];
    __shared__ __align__(16) char sB[2][TILE_BYTES];

    const int tid  = threadIdx.x;
    const int lane = tid & 31;
    const int wave = tid >> 5;
    const int m0 = blockIdx.x * BM;
    const int n0 = blockIdx.y * BN;
    const int wm = (wave >> 2) * 64;   // 2 waves along M
    const int wn = (wave & 3) * 32;    // 4 waves along N

    const v8f vzero = {0.f, 0.f, 0.f, 0.f, 0.f, 0.f, 0.f, 0.f};
    v8f acc[4][2];
#pragma unroll
    for (int i = 0; i < 4; ++i)
#pragma unroll
        for (int j = 0; j < 2; ++j) acc[i][j] = vzero;

    float4 ra[8], rb[8];
    gload_f32(X,  m0, HIDDEN, 0, ra, tid, true);
    gload_f32(W1, n0, HIDDEN, 0, rb, tid, true);

    for (int k0 = 0; k0 < HIDDEN; k0 += BK) {
        const int buf = (k0 / BK) & 1;
        lstore_f32(ra, sA[buf], tid);
        lstore_f32(rb, sB[buf], tid);
        __syncthreads();
        if (k0 + BK < HIDDEN) {                         // hide latency under WMMAs
            const bool pf = (k0 + 2 * BK) < HIDDEN;
            gload_f32(X,  m0, HIDDEN, k0 + BK, ra, tid, pf);
            gload_f32(W1, n0, HIDDEN, k0 + BK, rb, tid, pf);
        }
        mma_chunk(sA[buf], sB[buf], wm, wn, lane, acc);
    }

    // Epilogue: +bias, relu, bf16 store. C/D layout: lanes 0-15 -> rows v,
    // lanes 16-31 -> rows 8+v, N = lane&15.
    const int lhalf = ((lane >> 4) & 1) * 8;
    const int lcol  = lane & 15;
#pragma unroll
    for (int j = 0; j < 2; ++j) {
        const int n = n0 + wn + j * 16 + lcol;
        const float bias = b1[n];
#pragma unroll
        for (int i = 0; i < 4; ++i) {
            const int mbase = m0 + wm + i * 16 + lhalf;
#pragma unroll
            for (int v = 0; v < 8; ++v) {
                float val = acc[i][j][v] + bias;
                val = val > 0.f ? val : 0.f;
                act[(size_t)(mbase + v) * INTER + n] = f32_to_bf16_rne(val);
            }
        }
    }
}

// ---------------- GEMM2: Y = act @ W2^T + b2, f32 out ----------------------
__global__ __launch_bounds__(256) void ffn_gemm2(const unsigned short* __restrict__ act,
                                                 const float* __restrict__ W2,
                                                 const float* __restrict__ b2,
                                                 float* __restrict__ out) {
    __shared__ __align__(16) char sA[2][TILE_BYTES];
    __shared__ __align__(16) char sB[2][TILE_BYTES];

    const int tid  = threadIdx.x;
    const int lane = tid & 31;
    const int wave = tid >> 5;
    const int m0 = blockIdx.x * BM;
    const int n0 = blockIdx.y * BN;
    const int wm = (wave >> 2) * 64;
    const int wn = (wave & 3) * 32;

    const v8f vzero = {0.f, 0.f, 0.f, 0.f, 0.f, 0.f, 0.f, 0.f};
    v8f acc[4][2];
#pragma unroll
    for (int i = 0; i < 4; ++i)
#pragma unroll
        for (int j = 0; j < 2; ++j) acc[i][j] = vzero;

    uint4  ra[4];
    float4 rb[8];
    gload_bf16(act, m0, INTER, 0, ra, tid, true);
    gload_f32(W2,  n0, INTER, 0, rb, tid, true);

    for (int k0 = 0; k0 < INTER; k0 += BK) {
        const int buf = (k0 / BK) & 1;
        lstore_bf16(ra, sA[buf], tid);
        lstore_f32(rb, sB[buf], tid);
        __syncthreads();
        if (k0 + BK < INTER) {
            const bool pf = (k0 + 2 * BK) < INTER;
            gload_bf16(act, m0, INTER, k0 + BK, ra, tid, pf);
            gload_f32(W2,  n0, INTER, k0 + BK, rb, tid, pf);
        }
        mma_chunk(sA[buf], sB[buf], wm, wn, lane, acc);
    }

    const int lhalf = ((lane >> 4) & 1) * 8;
    const int lcol  = lane & 15;
#pragma unroll
    for (int j = 0; j < 2; ++j) {
        const int n = n0 + wn + j * 16 + lcol;
        const float bias = b2[n];
#pragma unroll
        for (int i = 0; i < 4; ++i) {
            const int mbase = m0 + wm + i * 16 + lhalf;
#pragma unroll
            for (int v = 0; v < 8; ++v)
                out[(size_t)(mbase + v) * HIDDEN + n] = acc[i][j][v] + bias;
        }
    }
}

extern "C" void kernel_launch(void* const* d_in, const int* in_sizes, int n_in,
                              void* d_out, int out_size, void* d_ws, size_t ws_size,
                              hipStream_t stream) {
    const float* X  = (const float*)d_in[0];   // [M, 1024]
    const float* W1 = (const float*)d_in[1];   // [4096, 1024]
    const float* b1 = (const float*)d_in[2];   // [4096]
    const float* W2 = (const float*)d_in[3];   // [1024, 4096]
    const float* b2 = (const float*)d_in[4];   // [1024]
    float* out = (float*)d_out;                // [M, 1024]

    const int M = in_sizes[0] / HIDDEN;        // 16384 for the given shapes

    // Workspace: bf16 activations, M*INTER*2 bytes (128 MB at M=16384).
    unsigned short* act = (unsigned short*)d_ws;

    dim3 block(256);
    dim3 grid1(M / BM, INTER / BN);            // 128 x 32 workgroups
    dim3 grid2(M / BM, HIDDEN / BN);           // 128 x 8 workgroups
    ffn_gemm1<<<grid1, block, 0, stream>>>(X, W1, b1, act);
    ffn_gemm2<<<grid2, block, 0, stream>>>(act, W2, b2, out);
}
